// WeightOnlyInt4Linear_2894807958009
// MI455X (gfx1250) — matrix operand order repaired and checked by reading, not compiled, not run on hardware
//
#include <hip/hip_runtime.h>

#define K_DIM   4096
#define O_DIM   11008
#define GSIZE   128
#define NGROUPS 32
#define BATCH   8

typedef __bf16 v16bf __attribute__((ext_vector_type(16)));
typedef __bf16 v8bf  __attribute__((ext_vector_type(8)));
typedef __bf16 v2bf  __attribute__((ext_vector_type(2)));
typedef float  v8f   __attribute__((ext_vector_type(8)));
typedef float  v4f   __attribute__((ext_vector_type(4)));
typedef float  v2f   __attribute__((ext_vector_type(2)));
typedef int    v4i   __attribute__((ext_vector_type(4)));

static __device__ __forceinline__ float bfbits2f(unsigned int bits) {
    unsigned int u = bits << 16;
    float f;
    __builtin_memcpy(&f, &u, 4);
    return f;
}
static __device__ __forceinline__ unsigned short f2bfbits(float f) {
    unsigned int u;
    __builtin_memcpy(&u, &f, 4);
    unsigned int r = u + 0x7FFFu + ((u >> 16) & 1u);   // round-to-nearest-even
    return (unsigned short)(r >> 16);
}

// One wave -> one 16-wide output tile over half of K. Wave pairs (kh=0/1)
// share a tile and reduce f32 accumulators through LDS.
__global__ __launch_bounds__(256)
void WeightOnlyInt4Linear_wmma_kernel(const __bf16* __restrict__ x,
                                      const int* __restrict__ qw,
                                      const unsigned short* __restrict__ saz,
                                      unsigned short* __restrict__ out)
{
    __shared__ float red[4 * 16 * 8];   // 4 tiles x 16 lanes x 8 f32

    const int tid   = threadIdx.x;
    const int lane  = tid & 31;
    const int wave  = tid >> 5;
    const int tile  = wave & 3;    // 4 output tiles per block
    const int kh    = wave >> 2;   // K-split half (0 or 1)
    const int nloc  = lane & 15;   // N within tile / M for A
    const int khalf = lane >> 4;   // upper/lower lane group

    const int obase = (blockIdx.x * 4 + tile) * 16;
    const int o     = obase + nloc;        // output feature this lane owns (B/N)
    const int m     = nloc & 7;            // batch row for A; pad rows mirror 0..7

    const __bf16* __restrict__ xrow = x + (size_t)m * K_DIM;
    const int*    __restrict__ qrow = qw + (size_t)o * K_DIM;

    v8f acc = {};

    const int g0 = kh * (NGROUPS / 2);
    for (int g = g0; g < g0 + NGROUPS / 2; ++g) {
        // scale (lo16) and zero (hi16) for this (group, out) in one b32 load
        unsigned int szbits =
            *(const unsigned int*)(saz + ((size_t)g * O_DIM + o) * 2);
        const float s  = bfbits2f(szbits & 0xFFFFu);
        const float z8 = __builtin_fmaf(s, -8.0f, bfbits2f(szbits >> 16)); // z - 8*s

#pragma unroll
        for (int c4 = 0; c4 < 4; ++c4) {
            const int kc = g * GSIZE + c4 * 32;

            // ---- A fragment: 16x32 bf16 of x (contiguous k order: 16*khalf + s) ----
            union { v16bf v; v8bf h[2]; } au;
            au.h[0] = *(const v8bf*)(xrow + kc + khalf * 16);       // K 0..7 (+16)
            au.h[1] = *(const v8bf*)(xrow + kc + 8 + khalf * 16);   // K 8..15 (+16)

            // ---- B fragment: 32x16 bf16 dequantized weights ----
            // lane holds K = kc + khalf*16 + i for its output column o
            const int* qp = qrow + kc + khalf * 16;
            union { v4i q[4]; int e[16]; } qi;
            qi.q[0] = __builtin_nontemporal_load((const v4i*)(qp + 0));
            qi.q[1] = __builtin_nontemporal_load((const v4i*)(qp + 4));
            qi.q[2] = __builtin_nontemporal_load((const v4i*)(qp + 8));
            qi.q[3] = __builtin_nontemporal_load((const v4i*)(qp + 12));

            // w = q*s + (z - 8*s); convert in pairs -> v_cvt_pk_bf16_f32
            union { v16bf v; v2bf p[8]; } bu;
#pragma unroll
            for (int j = 0; j < 8; ++j) {
                float w0 = __builtin_fmaf((float)qi.e[2 * j + 0], s, z8);
                float w1 = __builtin_fmaf((float)qi.e[2 * j + 1], s, z8);
                v2bf pk;
                pk.x = (__bf16)w0;
                pk.y = (__bf16)w1;
                bu.p[j] = pk;
            }

            acc = __builtin_amdgcn_wmma_f32_16x16x32_bf16(
                false, au.v, false, bu.v, (short)0, acc, false, false);
        }
    }

    // ---- cross-wave K reduction via LDS, then store bf16 ----
    union { v8f v; v4f h[2]; float e[8]; } cu;
    cu.v = acc;

    if (kh == 1 && lane < 16) {
        float* p = &red[(tile * 16 + lane) * 8];
        *(v4f*)(p + 0) = cu.h[0];
        *(v4f*)(p + 4) = cu.h[1];
    }
    __syncthreads();
    if (kh == 0 && lane < 16) {
        const float* p = &red[(tile * 16 + lane) * 8];
        cu.h[0] += *(const v4f*)(p + 0);
        cu.h[1] += *(const v4f*)(p + 4);
        // C/D layout: lanes 0..15, VGPR j -> M=j (batch row), N=lane
#pragma unroll
        for (int j = 0; j < BATCH; ++j) {
            out[(size_t)j * O_DIM + obase + lane] = f2bfbits(cu.e[j]);
        }
    }
}

extern "C" void kernel_launch(void* const* d_in, const int* in_sizes, int n_in,
                              void* d_out, int out_size, void* d_ws, size_t ws_size,
                              hipStream_t stream) {
    (void)in_sizes; (void)n_in; (void)out_size; (void)d_ws; (void)ws_size;
    const __bf16*         x   = (const __bf16*)d_in[0];
    const int*            qw  = (const int*)d_in[1];
    const unsigned short* saz = (const unsigned short*)d_in[2];
    unsigned short*       out = (unsigned short*)d_out;

    dim3 grid(O_DIM / 64);   // 172 blocks: 4 tiles/block * 16 outputs/tile
    dim3 block(256);         // 8 waves: 4 tiles x 2 K-halves
    hipLaunchKernelGGL(WeightOnlyInt4Linear_wmma_kernel, grid, block, 0, stream,
                       x, qw, saz, out);
}
